// RRoiPosFeatExtraxtor_49443663511570
// MI455X (gfx1250) — matrix operand order repaired and checked by reading, not compiled, not run on hardware
//
#include <hip/hip_runtime.h>
#include <hip/hip_bf16.h>
#include <cmath>

typedef __attribute__((ext_vector_type(16))) _Float16 v16h;
typedef __attribute__((ext_vector_type(8)))  _Float16 v8h;
typedef __attribute__((ext_vector_type(8)))  float    v8f;

#define PI_F 3.14159265358979323846f

// ---------------------------------------------------------------------------
// 1) NCHW -> NHWC transpose (tiled through LDS, coalesced both sides)
// ---------------------------------------------------------------------------
__global__ __launch_bounds__(256) void nchw_to_nhwc(const float* __restrict__ in,
                                                    float* __restrict__ out,
                                                    int C, long HW) {
  __shared__ float tile[32][33];
  const int b  = blockIdx.z;
  const long hw0 = (long)blockIdx.x * 32;
  const int  c0  = blockIdx.y * 32;
  const int tx = threadIdx.x & 31;
  const int ty = threadIdx.x >> 5;               // 0..7
  const float* src = in + (size_t)b * C * HW;
  float* dst = out + (size_t)b * HW * C;
#pragma unroll
  for (int i = 0; i < 32; i += 8)
    tile[ty + i][tx] = src[(size_t)(c0 + ty + i) * HW + hw0 + tx];
  __syncthreads();
#pragma unroll
  for (int i = 0; i < 32; i += 8)
    dst[(size_t)(hw0 + ty + i) * C + c0 + tx] = tile[tx][ty + i];
}

// ---------------------------------------------------------------------------
// 2) poly2obb + obb2poly per ROI (tiny)
// ---------------------------------------------------------------------------
__global__ void prep_boxes(const float* __restrict__ boxes,
                           float* __restrict__ obb,
                           float* __restrict__ polys, int R) {
  int i = blockIdx.x * blockDim.x + threadIdx.x;
  if (i >= R) return;
  const float* p = boxes + (size_t)i * 8;
  float x1 = p[0], y1 = p[1], x2 = p[2], y2 = p[3], x3 = p[4], y3 = p[5];
  float dx1 = x2 - x1, dy1 = y2 - y1, dx2 = x3 - x2, dy2 = y3 - y2;
  float e1 = sqrtf(dx1 * dx1 + dy1 * dy1);
  float e2 = sqrtf(dx2 * dx2 + dy2 * dy2);
  float a1 = atan2f(dy1, dx1);
  float a2 = atan2f(dy2, dx2);
  float ang = (e1 >= e2) ? a1 : a2;
  float t = ang + 0.5f * PI_F;
  t = t - floorf(t / PI_F) * PI_F;               // jnp.mod semantics (mod>0)
  ang = t - 0.5f * PI_F;
  float w = fmaxf(e1, e2), h = fminf(e1, e2);
  float cx = 0.5f * (x1 + x3), cy = 0.5f * (y1 + y3);
  obb[i * 5 + 0] = cx; obb[i * 5 + 1] = cy;
  obb[i * 5 + 2] = w;  obb[i * 5 + 3] = h; obb[i * 5 + 4] = ang;
  float c = cosf(ang), s = sinf(ang);
  float v1x = 0.5f * w * c, v1y = 0.5f * w * s;
  float v2x = -0.5f * h * s, v2y = 0.5f * h * c;
  float* q = polys + (size_t)i * 8;
  q[0] = cx + v1x + v2x; q[1] = cy + v1y + v2y;
  q[2] = cx + v1x - v2x; q[3] = cy + v1y - v2y;
  q[4] = cx - v1x - v2x; q[5] = cy - v1y - v2y;
  q[6] = cx - v1x + v2x; q[7] = cy - v1y + v2y;
}

// ---------------------------------------------------------------------------
// 3) Rotated ROI-align on NHWC feats -> f16 GEMM A matrix [R][bin*C + c]
// ---------------------------------------------------------------------------
__global__ __launch_bounds__(256) void roi_align_rot(const float* __restrict__ nhwc,
                                                     const float* __restrict__ obb,
                                                     _Float16* __restrict__ xA,
                                                     int C, int H, int W,
                                                     int NperB, int Fin) {
  const int blk = blockIdx.x;
  const int roi = blk / 9;
  const int bin = blk % 9;
  const int py = bin / 3, px = bin % 3;
  const int c = threadIdx.x;
  const int b = roi / NperB;
  const float cx = obb[roi * 5 + 0] * 0.25f - 0.5f;
  const float cy = obb[roi * 5 + 1] * 0.25f - 0.5f;
  const float rw = obb[roi * 5 + 2] * 0.25f;
  const float rh = obb[roi * 5 + 3] * 0.25f;
  const float th = obb[roi * 5 + 4];
  const float binh = rh * (1.0f / 3.0f), binw = rw * (1.0f / 3.0f);
  const float ct = cosf(th), st = sinf(th);
  const float* base = nhwc + (size_t)b * H * W * C;
  float acc = 0.f;
#pragma unroll
  for (int sy = 0; sy < 2; ++sy)
#pragma unroll
    for (int sx = 0; sx < 2; ++sx) {
      float gy = (float)py + (sy + 0.5f) * 0.5f;
      float gx = (float)px + (sx + 0.5f) * 0.5f;
      float yy = -rh * 0.5f + binh * gy;
      float xx = -rw * 0.5f + binw * gx;
      float x = xx * ct - yy * st + cx;
      float y = xx * st + yy * ct + cy;
      bool ok = (y > -1.f) && (y < (float)H) && (x > -1.f) && (x < (float)W);
      y = fminf(fmaxf(y, 0.f), (float)(H - 1));
      x = fminf(fmaxf(x, 0.f), (float)(W - 1));
      float y0f = floorf(y), x0f = floorf(x);
      int y0 = (int)y0f, x0 = (int)x0f;
      int y1 = min(y0 + 1, H - 1), x1 = min(x0 + 1, W - 1);
      float ly = y - y0f, lx = x - x0f, hy = 1.f - ly, hx = 1.f - lx;
      float v00 = base[((size_t)y0 * W + x0) * C + c];
      float v01 = base[((size_t)y0 * W + x1) * C + c];
      float v10 = base[((size_t)y1 * W + x0) * C + c];
      float v11 = base[((size_t)y1 * W + x1) * C + c];
      float v = hy * hx * v00 + hy * lx * v01 + ly * hx * v10 + ly * lx * v11;
      acc += ok ? v : 0.f;
    }
  xA[(size_t)roi * Fin + bin * C + c] = (_Float16)(acc * 0.25f);
}

// ---------------------------------------------------------------------------
// 4) Weight pack: fp32 K x N row-major -> f16 slab layout [kb][n][32]
//    (optionally remapping K: dest k = bin*C + c  <- src k = c*9 + bin)
//    Slab layout makes the GEMM's B staging a single contiguous 4KB copy and
//    every lane's B fragment two b128 LDS loads.
// ---------------------------------------------------------------------------
__global__ __launch_bounds__(256) void pack_w(const float* __restrict__ in,
                                              _Float16* __restrict__ out,
                                              int N, int C, int perm) {
  __shared__ float tile[32][65];
  const int kb = blockIdx.x;
  const int n0 = blockIdx.y * 64;
  const int t = threadIdx.x;
  const int tn = t & 63, tk = t >> 6;            // tk 0..3
#pragma unroll
  for (int i = 0; i < 32; i += 4) {
    int kk = tk + i;
    int kd = kb * 32 + kk;                       // dest feature index
    int ks = perm ? ((kd % C) * 9 + kd / C) : kd;
    tile[kk][tn] = in[(size_t)ks * N + n0 + tn];
  }
  __syncthreads();
  const int n_l = t >> 2, kk0 = (t & 3) * 8;
  _Float16 tmp[8];
#pragma unroll
  for (int j = 0; j < 8; ++j) tmp[j] = (_Float16)tile[kk0 + j][n_l];
  size_t base = ((size_t)kb * N + n0) * 32;
  *(v8h*)&out[base + (size_t)n_l * 32 + kk0] = *(const v8h*)tmp;
}

// ---------------------------------------------------------------------------
// 5) WMMA GEMM: 128x64 block tile, 8 waves, each wave = 16x64 slab (4 accums).
//    A staged in LDS [row][32]; B staged in LDS [n][32] (from packed slabs).
//    All fragment loads are b128; 4 x v_wmma_f32_16x16x32_f16 per stage/wave.
// ---------------------------------------------------------------------------
template <bool RELU, bool OUT_F16>
__global__ __launch_bounds__(256) void gemm_wmma(const _Float16* __restrict__ A,
                                                 const _Float16* __restrict__ Bw,
                                                 const float* __restrict__ bias,
                                                 void* __restrict__ Cout,
                                                 int M, int N, int K) {
  constexpr int BM = 128, BN = 64, KB = 32;
  __shared__ __align__(16) _Float16 As[BM * KB];   // 8 KB
  __shared__ __align__(16) _Float16 Bs[BN * KB];   // 4 KB
  const int tid  = threadIdx.x;
  const int lane = tid & 31;
  const int wave = tid >> 5;                       // wave = m-tile (0..7)
  const int bm = blockIdx.y * BM;
  const int bn = blockIdx.x * BN;
  const int mloc   = lane & 15;
  const int hiHalf = lane >> 4;
  const int arow = tid >> 1, acol = (tid & 1) * 16;

  v8f acc0 = {}, acc1 = {}, acc2 = {}, acc3 = {};
  const int nkb = K / KB;

  for (int kb = 0; kb < nkb; ++kb) {
    const int k0 = kb * KB;
    const _Float16* ag = &A[(size_t)(bm + arow) * K + k0 + acol];
    *(v8h*)&As[arow * KB + acol]     = *(const v8h*)ag;
    *(v8h*)&As[arow * KB + acol + 8] = *(const v8h*)(ag + 8);
    const v8h* bg = (const v8h*)&Bw[((size_t)kb * N + bn) * KB];
    ((v8h*)Bs)[tid] = bg[tid];
    if (kb + 1 < nkb) {                            // global_prefetch_b8 next tiles
      __builtin_prefetch(ag + KB, 0, 3);
      __builtin_prefetch((const _Float16*)bg + (size_t)N * KB, 0, 3);
    }
    __syncthreads();

    // A fragment (ISA 16-bit A 16x32): two contiguous 8-half runs per lane
    const _Float16* arow_p = &As[(wave * 16 + mloc) * KB];
    v8h alo = *(const v8h*)(arow_p + hiHalf * 8);
    v8h ahi = *(const v8h*)(arow_p + 16 + hiHalf * 8);
    v16h af = __builtin_shufflevector(alo, ahi, 0, 1, 2, 3, 4, 5, 6, 7,
                                                8, 9, 10, 11, 12, 13, 14, 15);
#pragma unroll
    for (int nt = 0; nt < 4; ++nt) {
      const _Float16* brow_p = &Bs[(nt * 16 + mloc) * KB];
      v8h blo = *(const v8h*)(brow_p + hiHalf * 8);
      v8h bhi = *(const v8h*)(brow_p + 16 + hiHalf * 8);
      v16h bf = __builtin_shufflevector(blo, bhi, 0, 1, 2, 3, 4, 5, 6, 7,
                                                  8, 9, 10, 11, 12, 13, 14, 15);
      switch (nt) {
        case 0: acc0 = __builtin_amdgcn_wmma_f32_16x16x32_f16(false, af, false, bf, (short)0, acc0, false, false); break;
        case 1: acc1 = __builtin_amdgcn_wmma_f32_16x16x32_f16(false, af, false, bf, (short)0, acc1, false, false); break;
        case 2: acc2 = __builtin_amdgcn_wmma_f32_16x16x32_f16(false, af, false, bf, (short)0, acc2, false, false); break;
        case 3: acc3 = __builtin_amdgcn_wmma_f32_16x16x32_f16(false, af, false, bf, (short)0, acc3, false, false); break;
      }
    }
    __syncthreads();
  }

  // Epilogue: C/D layout -> M = r + 8*hiHalf, N = lane%16
  const int mAdd = hiHalf * 8;
  v8f* accs[4] = {&acc0, &acc1, &acc2, &acc3};
#pragma unroll
  for (int nt = 0; nt < 4; ++nt) {
    int gn = bn + nt * 16 + mloc;
    float bv = bias[gn];
#pragma unroll
    for (int r = 0; r < 8; ++r) {
      int gm = bm + wave * 16 + r + mAdd;
      float v = (*accs[nt])[r] + bv;
      if (RELU) v = fmaxf(v, 0.f);
      if (OUT_F16) ((_Float16*)Cout)[(size_t)gm * N + gn] = (_Float16)v;
      else         ((float*)Cout)[(size_t)gm * N + gn] = v;
    }
  }
}

// ---------------------------------------------------------------------------
// 6) Fused bbox projection + LN + residual + LN (one block per row, D=512)
//    wave32 shuffle reduction + 8-partial LDS combine
// ---------------------------------------------------------------------------
__device__ __forceinline__ float blk_sum(float v, float* part) {
#pragma unroll
  for (int o = 16; o > 0; o >>= 1) v += __shfl_xor(v, o, 32);
  const int lane = threadIdx.x & 31, w = threadIdx.x >> 5;
  __syncthreads();
  if (lane == 0) part[w] = v;
  __syncthreads();
  float r = 0.f;
#pragma unroll
  for (int j = 0; j < 8; ++j) r += part[j];
  return r;
}

__global__ __launch_bounds__(256) void final_ln(const float* __restrict__ x2,
                                                const float* __restrict__ polys,
                                                const float* __restrict__ img_sizes,
                                                const float* __restrict__ bboxW,
                                                const float* __restrict__ bboxb,
                                                const float* __restrict__ g_bln,
                                                const float* __restrict__ b_bln,
                                                const float* __restrict__ g_aln,
                                                const float* __restrict__ b_aln,
                                                float* __restrict__ out,
                                                int D, int NperB) {
  __shared__ float ps[8];
  __shared__ float part[8];
  const int row = blockIdx.x;
  const int b = row / NperB;
  const int t = threadIdx.x;
  if (t < 8) ps[t] = polys[(size_t)row * 8 + t] / img_sizes[b * 2 + (t & 1)];
  __syncthreads();
  const int d0 = t, d1 = t + 256;
  float p0 = bboxb[d0], p1 = bboxb[d1];
#pragma unroll
  for (int j = 0; j < 8; ++j) {
    p0 += ps[j] * bboxW[(size_t)j * D + d0];
    p1 += ps[j] * bboxW[(size_t)j * D + d1];
  }
  float s  = blk_sum(p0 + p1, part);
  float s2 = blk_sum(p0 * p0 + p1 * p1, part);
  float mu  = s / (float)D;
  float var = s2 / (float)D - mu * mu;
  float inv = rsqrtf(var + 1e-5f);
  float bb0 = (p0 - mu) * inv * g_bln[d0] + b_bln[d0];
  float bb1 = (p1 - mu) * inv * g_bln[d1] + b_bln[d1];
  float y0 = x2[(size_t)row * D + d0] + bb0;
  float y1 = x2[(size_t)row * D + d1] + bb1;
  s  = blk_sum(y0 + y1, part);
  s2 = blk_sum(y0 * y0 + y1 * y1, part);
  mu  = s / (float)D;
  var = s2 / (float)D - mu * mu;
  inv = rsqrtf(var + 1e-5f);
  out[(size_t)row * D + d0] = (y0 - mu) * inv * g_aln[d0] + b_aln[d0];
  out[(size_t)row * D + d1] = (y1 - mu) * inv * g_aln[d1] + b_aln[d1];
}

// ---------------------------------------------------------------------------
// Host driver
// ---------------------------------------------------------------------------
extern "C" void kernel_launch(void* const* d_in, const int* in_sizes, int n_in,
                              void* d_out, int out_size, void* d_ws, size_t ws_size,
                              hipStream_t stream) {
  const float* feats     = (const float*)d_in[0];
  const float* boxes     = (const float*)d_in[1];
  const float* img_sizes = (const float*)d_in[2];
  const float* W1        = (const float*)d_in[3];
  const float* b1        = (const float*)d_in[4];
  const float* W2        = (const float*)d_in[5];
  const float* b2        = (const float*)d_in[6];
  const float* bboxW     = (const float*)d_in[7];
  const float* bboxb     = (const float*)d_in[8];
  const float* g_bln     = (const float*)d_in[9];
  const float* b_bln     = (const float*)d_in[10];
  const float* g_aln     = (const float*)d_in[11];
  const float* b_aln     = (const float*)d_in[12];
  float* out = (float*)d_out;

  const int  D     = in_sizes[4];            // 512
  const int  R     = in_sizes[1] / 8;        // 1024
  const int  B     = in_sizes[2] / 2;        // 2
  const int  NperB = R / B;
  const int  Fin   = in_sizes[3] / D;        // 2304
  const int  C     = Fin / 9;                // 256
  const long HW    = (long)in_sizes[0] / B / C;
  const int  H     = (int)(sqrt((double)HW) + 0.5);
  const int  W     = H;

  char* wsb = (char*)d_ws;
  size_t off = 0;
  auto take = [&](size_t bytes) -> void* {
    void* p = wsb + off;
    off += (bytes + 255) & ~(size_t)255;
    return p;
  };
  float*     nhwc   = (float*)take((size_t)B * HW * C * sizeof(float));
  float*     obbw   = (float*)take((size_t)R * 5 * sizeof(float));
  float*     polysw = (float*)take((size_t)R * 8 * sizeof(float));
  _Float16*  xA     = (_Float16*)take((size_t)R * Fin * sizeof(_Float16));
  _Float16*  W1p    = (_Float16*)take((size_t)Fin * D * sizeof(_Float16));
  _Float16*  W2p    = (_Float16*)take((size_t)D * D * sizeof(_Float16));
  _Float16*  h1     = (_Float16*)take((size_t)R * D * sizeof(_Float16));
  float*     x2     = (float*)take((size_t)R * D * sizeof(float));

  // 1) NCHW -> NHWC (one pass; makes every ROI bilinear tap lane-coalesced)
  nchw_to_nhwc<<<dim3((unsigned)(HW / 32), C / 32, B), 256, 0, stream>>>(feats, nhwc, C, HW);

  // 2) box math
  prep_boxes<<<(R + 255) / 256, 256, 0, stream>>>(boxes, obbw, polysw, R);

  // 3) weight packing into WMMA slab layout (W1 rows permuted to bin-major)
  pack_w<<<dim3(Fin / 32, D / 64), 256, 0, stream>>>(W1, W1p, D, C, 1);
  pack_w<<<dim3(D / 32,  D / 64), 256, 0, stream>>>(W2, W2p, D, C, 0);

  // 4) rotated ROI align -> f16 A matrix
  roi_align_rot<<<R * 9, C, 0, stream>>>(nhwc, obbw, xA, C, H, W, NperB, Fin);

  // 5) MLP on WMMA: relu(xA @ W1 + b1) -> h1 (f16);  h1 @ W2 + b2 -> x2 (f32)
  gemm_wmma<true,  true ><<<dim3(D / 64, R / 128), 256, 0, stream>>>(xA, W1p, b1, (void*)h1, R, D, Fin);
  gemm_wmma<false, false><<<dim3(D / 64, R / 128), 256, 0, stream>>>(h1, W2p, b2, (void*)x2, R, D, D);

  // 6) bbox branch + double LayerNorm, fused
  final_ln<<<R, 256, 0, stream>>>(x2, polysw, img_sizes, bboxW, bboxb,
                                  g_bln, b_bln, g_aln, b_aln, out, D, NperB);
}
